// MultiHeadedAttention_73564199846301
// MI455X (gfx1250) — compile-verified, matrix operand order checked by reading, not run on hardware
//
#include <hip/hip_runtime.h>
#include <stdint.h>

// ---------------------------------------------------------------------------
// MultiHeadedAttention for MI455X (gfx1250): bf16 WMMA + async global->LDS
// double-buffered staging. B=2, S=4096, D=768, H=12, DK=64.
// ---------------------------------------------------------------------------

typedef __attribute__((ext_vector_type(16))) __bf16       v16bf;
typedef __attribute__((ext_vector_type(8)))  float        v8f;
typedef __attribute__((ext_vector_type(4)))  unsigned int u32x4;

#define B_    2
#define S_    4096
#define D_    768
#define H_    12
#define DK_   64
#define MTOT  (B_ * S_)          // 8192 tokens

__device__ __forceinline__ unsigned short f2bf(float f) {
    unsigned int u = __float_as_uint(f);
    u += 0x7FFFu + ((u >> 16) & 1u);       // round-to-nearest-even
    return (unsigned short)(u >> 16);
}

union Frag {                                // one 16x32 bf16 WMMA operand
    v16bf          v;
    u32x4          q[2];
    unsigned short s[16];
};

__device__ __forceinline__ void load_frag(Frag& f, const unsigned short* p) {
    f.q[0] = *(const u32x4*)(p);           // 16B
    f.q[1] = *(const u32x4*)(p + 8);       // 16B  (32B contiguous per lane)
}

// Low 32 bits of a flat pointer into __shared__ == LDS byte offset (ISA 10.2).
__device__ __forceinline__ unsigned lds_off32(const void* p) {
    return (unsigned)(unsigned long long)(uintptr_t)p;
}

// CDNA5 async global->LDS copy, 16B per lane, tracked by ASYNCcnt.
__device__ __forceinline__ void async_b128(const void* dst_lds, const void* src) {
    asm volatile("global_load_async_to_lds_b128 %0, %1, off"
                 :: "v"(lds_off32(dst_lds)),
                    "v"((unsigned long long)(uintptr_t)src)
                 : "memory");
}
__device__ __forceinline__ void wait_async() {
    asm volatile("s_wait_asynccnt 0x0" ::: "memory");
}

// ---------------------------------------------------------------------------
// fp32 -> bf16 bulk conversion
// ---------------------------------------------------------------------------
__global__ void cvt_f32_bf16(const float* __restrict__ src,
                             unsigned short* __restrict__ dst, int n) {
    int i = blockIdx.x * blockDim.x + threadIdx.x;
    if (i < n) dst[i] = f2bf(src[i]);
}

// ---------------------------------------------------------------------------
// Tiled WMMA GEMM:  Y[M,N] = A[M,768] * W[N,768]^T + bias
//   mode 0: Y -> Qh  [B,H,S,DK] bf16
//   mode 1: Y -> Kh  [B,H,S,DK] bf16
//   mode 2: Y -> Vt  [B,H,DK,S] bf16 (transposed for PV B-fragments)
//   mode 3: Y -> outf [M,768] fp32 (final projection)
// Block: 256 threads = 8 waves (2 x 4), block tile 128x128, wave tile 64x32.
// Tiles staged with async global->LDS copies, double buffered.
// ---------------------------------------------------------------------------
__global__ __launch_bounds__(256) void gemm_wmma(
        const unsigned short* __restrict__ A,
        const unsigned short* __restrict__ W,
        const float*          __restrict__ bias,
        unsigned short*       __restrict__ outb,
        float*                __restrict__ outf,
        int mode)
{
    __shared__ unsigned short sa[2][128 * 40];   // 32 cols + 8 pad (80B rows)
    __shared__ unsigned short sb[2][128 * 40];

    const int tid  = threadIdx.x;
    const int lane = tid & 31;
    const int wave = tid >> 5;
    const int wm   = wave & 1;       // 2 M sub-tiles of 64
    const int wn   = wave >> 1;      // 4 N sub-tiles of 32
    const int m0   = blockIdx.x * 128;
    const int n0   = blockIdx.y * 128;

    const v8f zero = {};
    v8f acc[4][2];
#pragma unroll
    for (int i = 0; i < 4; ++i)
#pragma unroll
        for (int j = 0; j < 2; ++j) acc[i][j] = zero;

    // issue async copies for one 128x32 A tile + 128x32 W tile (16KB total)
    auto stage = [&](int kb, int buf) {
#pragma unroll
        for (int r2 = 0; r2 < 2; ++r2) {
            const int c   = tid + r2 * 256;     // chunk of 8 bf16 (16B)
            const int e   = c * 8;
            const int row = e >> 5;             // 0..127
            const int col = e & 31;             // multiple of 8
            async_b128(&sa[buf][row * 40 + col],
                       A + (size_t)(m0 + row) * D_ + kb + col);
            async_b128(&sb[buf][row * 40 + col],
                       W + (size_t)(n0 + row) * D_ + kb + col);
        }
    };

    stage(0, 0);
    wait_async();
    __syncthreads();

    const int koff = (lane >> 4) * 16;   // A/B lane K-half selector
    const int l15  = lane & 15;

    int cur = 0;
    for (int kb = 0; kb < D_; kb += 32, cur ^= 1) {
        if (kb + 32 < D_) stage(kb + 32, cur ^ 1);   // overlap copy w/ math
        if (kb + 64 < D_) {                          // L2 prefetch 2 ahead
            __builtin_prefetch(A + (size_t)(m0 + (tid >> 1)) * D_ + kb + 64, 0, 1);
            __builtin_prefetch(W + (size_t)(n0 + (tid >> 1)) * D_ + kb + 64, 0, 1);
        }

        Frag afr[4], bfr[2];
#pragma unroll
        for (int mt = 0; mt < 4; ++mt)
            load_frag(afr[mt], &sa[cur][(wm * 64 + mt * 16 + l15) * 40 + koff]);
#pragma unroll
        for (int nt = 0; nt < 2; ++nt)
            load_frag(bfr[nt], &sb[cur][(wn * 32 + nt * 16 + l15) * 40 + koff]);
#pragma unroll
        for (int mt = 0; mt < 4; ++mt)
#pragma unroll
            for (int nt = 0; nt < 2; ++nt)
                acc[mt][nt] = __builtin_amdgcn_wmma_f32_16x16x32_bf16(
                    false, afr[mt].v, false, bfr[nt].v,
                    (short)0, acc[mt][nt], false, false);

        wait_async();       // next-buffer copies done (own wave's ASYNCcnt)
        __syncthreads();    // everyone finished reading cur buffer
    }

    // ---- epilogue: bias + layout-specific store ----
    const int hi8 = (lane >> 4) * 8;     // C layout: lanes 16-31 hold M+8
#pragma unroll
    for (int nt = 0; nt < 2; ++nt) {
        const int n  = n0 + wn * 32 + nt * 16 + (lane & 15);
        const float bv = bias[n];
        const int h  = n >> 6;           // n / DK
        const int nd = n & 63;           // n % DK
#pragma unroll
        for (int mt = 0; mt < 4; ++mt) {
#pragma unroll
            for (int r = 0; r < 8; ++r) {
                const int   m   = m0 + wm * 64 + mt * 16 + r + hi8;
                const float val = acc[mt][nt][r] + bv;
                if (mode == 3) {
                    outf[(size_t)m * D_ + n] = val;
                } else {
                    const int bb = m >> 12;     // m / S
                    const int s  = m & 4095;    // m % S
                    size_t addr;
                    if (mode == 2)              // V transposed [B,H,DK,S]
                        addr = (((size_t)(bb * H_ + h)) * DK_ + nd) * S_ + s;
                    else                        // Q/K [B,H,S,DK]
                        addr = (((size_t)(bb * H_ + h)) * S_ + s) * DK_ + nd;
                    outb[addr] = f2bf(val);
                }
            }
        }
    }
}

// ---------------------------------------------------------------------------
// Flash attention: 4 waves / block, each wave owns 32 query rows (128/block).
// K (32x64) and V^T (64x32) tiles staged once per key-step via async copies
// (double buffered) and shared by all 4 waves.
// Per 32-key step / wave: 8 WMMAs (QK^T) + online softmax + 8 WMMAs (PV).
// ---------------------------------------------------------------------------
__global__ __launch_bounds__(128) void attn_wmma(
        const unsigned short* __restrict__ Qh,
        const unsigned short* __restrict__ Kh,
        const unsigned short* __restrict__ Vt,
        const int*            __restrict__ mask,
        unsigned short*       __restrict__ Oc)
{
    __shared__ unsigned short skt[2][32 * 72];   // keys x dims (+8 pad)
    __shared__ unsigned short svt[2][64 * 40];   // dims x keys (+8 pad)
    __shared__ unsigned short spt[4][32 * 40];   // per-wave P tile

    const int tid  = threadIdx.x;
    const int lane = tid & 31;
    const int wave = tid >> 5;
    const int qb   = blockIdx.x * 128 + wave * 32;
    const int h    = blockIdx.y;
    const int b    = blockIdx.z;

    const unsigned short* Q = Qh + ((size_t)(b * H_ + h)) * S_ * DK_;
    const unsigned short* K = Kh + ((size_t)(b * H_ + h)) * S_ * DK_;
    const unsigned short* V = Vt + ((size_t)(b * H_ + h)) * DK_ * S_;
    const int* mbase        = mask + (size_t)b * S_ * S_;

    const int l15  = lane & 15;
    const int hi8  = (lane >> 4) * 8;
    const int ksel = (lane >> 4) * 16;

    // stage one K tile + one V^T tile (4KB each) across the 128 threads
    auto stage = [&](int kb, int buf) {
#pragma unroll
        for (int r2 = 0; r2 < 2; ++r2) {
            const int c  = tid + r2 * 128;   // chunk of 8 bf16 (16B)
            const int e  = c * 8;
            const int kr = e >> 6, kc = e & 63;   // K tile 32x64
            async_b128(&skt[buf][kr * 72 + kc],
                       K + (size_t)(kb + kr) * DK_ + kc);
            const int vr = e >> 5, vc = e & 31;   // V^T tile 64x32
            async_b128(&svt[buf][vr * 40 + vc],
                       V + (size_t)vr * S_ + kb + vc);
        }
    };

    // Q fragments live in registers for the whole pass: 2 M-tiles x 2 K-tiles
    Frag qa[2][2];
#pragma unroll
    for (int mt = 0; mt < 2; ++mt)
#pragma unroll
        for (int kk = 0; kk < 2; ++kk)
            load_frag(qa[mt][kk],
                      Q + (size_t)(qb + mt * 16 + l15) * DK_ + kk * 32 + ksel);

    const v8f zero = {};
    v8f accO[2][4];
#pragma unroll
    for (int mt = 0; mt < 2; ++mt)
#pragma unroll
        for (int dt = 0; dt < 4; ++dt) accO[mt][dt] = zero;

    float mst[2][8], lst[2][8];
#pragma unroll
    for (int mt = 0; mt < 2; ++mt)
#pragma unroll
        for (int r = 0; r < 8; ++r) { mst[mt][r] = -1e30f; lst[mt][r] = 0.0f; }

    stage(0, 0);
    wait_async();
    __syncthreads();

    int cur = 0;
    for (int kb = 0; kb < S_; kb += 32, cur ^= 1) {
        if (kb + 32 < S_) stage(kb + 32, cur ^ 1);   // overlap copy w/ math

        // ---------- scores: S = Q * K^T (K fragments from LDS) ----------
        v8f sc[2][2];
#pragma unroll
        for (int mt = 0; mt < 2; ++mt)
#pragma unroll
            for (int nt = 0; nt < 2; ++nt) sc[mt][nt] = zero;

        Frag kf[2][2];   // [nt][kk]
#pragma unroll
        for (int nt = 0; nt < 2; ++nt)
#pragma unroll
            for (int kk = 0; kk < 2; ++kk)
                load_frag(kf[nt][kk],
                          &skt[cur][(nt * 16 + l15) * 72 + kk * 32 + ksel]);

#pragma unroll
        for (int mt = 0; mt < 2; ++mt)
#pragma unroll
            for (int nt = 0; nt < 2; ++nt)
#pragma unroll
                for (int kk = 0; kk < 2; ++kk)
                    sc[mt][nt] = __builtin_amdgcn_wmma_f32_16x16x32_bf16(
                        false, qa[mt][kk].v, false, kf[nt][kk].v,
                        (short)0, sc[mt][nt], false, false);

        // ---------- scale + mask ----------
#pragma unroll
        for (int mt = 0; mt < 2; ++mt)
#pragma unroll
            for (int nt = 0; nt < 2; ++nt)
#pragma unroll
                for (int r = 0; r < 8; ++r) {
                    float sv = sc[mt][nt][r] * 0.125f;   // 1/sqrt(64)
                    const int row = qb + mt * 16 + r + hi8;
                    const int col = kb + nt * 16 + l15;
                    if (mbase[(size_t)row * S_ + col] == 0) sv = -1e9f;
                    sc[mt][nt][r] = sv;
                }

        // ---------- online softmax (row reductions across 16-lane halves) --
#pragma unroll
        for (int mt = 0; mt < 2; ++mt)
#pragma unroll
            for (int r = 0; r < 8; ++r) {
                float mx = fmaxf(sc[mt][0][r], sc[mt][1][r]);
#pragma unroll
                for (int off = 8; off >= 1; off >>= 1)
                    mx = fmaxf(mx, __shfl_xor(mx, off, 32));
                const float mn    = fmaxf(mst[mt][r], mx);
                const float alpha = __expf(mst[mt][r] - mn);
                mst[mt][r] = mn;
                const float p0 = __expf(sc[mt][0][r] - mn);
                const float p1 = __expf(sc[mt][1][r] - mn);
                sc[mt][0][r] = p0;
                sc[mt][1][r] = p1;
                float rs = p0 + p1;
#pragma unroll
                for (int off = 8; off >= 1; off >>= 1)
                    rs += __shfl_xor(rs, off, 32);
                lst[mt][r] = lst[mt][r] * alpha + rs;
#pragma unroll
                for (int dt = 0; dt < 4; ++dt) accO[mt][dt][r] *= alpha;
            }

        // ---------- P: C-layout -> LDS -> A-fragments ----------
#pragma unroll
        for (int mt = 0; mt < 2; ++mt)
#pragma unroll
            for (int nt = 0; nt < 2; ++nt)
#pragma unroll
                for (int r = 0; r < 8; ++r)
                    spt[wave][(mt * 16 + r + hi8) * 40 + nt * 16 + l15] =
                        f2bf(sc[mt][nt][r]);
        __syncthreads();   // uniform across waves; also orders own ds stores

        Frag pa[2], vf[4];
#pragma unroll
        for (int mt = 0; mt < 2; ++mt)
            load_frag(pa[mt], &spt[wave][(mt * 16 + l15) * 40 + ksel]);
#pragma unroll
        for (int dt = 0; dt < 4; ++dt)
            load_frag(vf[dt], &svt[cur][(dt * 16 + l15) * 40 + ksel]);

#pragma unroll
        for (int mt = 0; mt < 2; ++mt)
#pragma unroll
            for (int dt = 0; dt < 4; ++dt)
                accO[mt][dt] = __builtin_amdgcn_wmma_f32_16x16x32_bf16(
                    false, pa[mt].v, false, vf[dt].v,
                    (short)0, accO[mt][dt], false, false);

        wait_async();       // next-buffer K/V copies complete
        __syncthreads();    // all waves done reading cur buffer
    }

    // ---------- normalize + store merged-head O [B,S,H*DK] bf16 ----------
#pragma unroll
    for (int mt = 0; mt < 2; ++mt)
#pragma unroll
        for (int r = 0; r < 8; ++r) {
            const float inv = 1.0f / lst[mt][r];
            const int   tok = qb + mt * 16 + r + hi8;
#pragma unroll
            for (int dt = 0; dt < 4; ++dt) {
                const int col = h * DK_ + dt * 16 + l15;
                Oc[((size_t)(b * S_ + tok)) * D_ + col] =
                    f2bf(accO[mt][dt][r] * inv);
            }
        }
}

// ---------------------------------------------------------------------------
// Launch: cvt -> Q/K/V GEMMs -> flash attention -> output projection
// ---------------------------------------------------------------------------
extern "C" void kernel_launch(void* const* d_in, const int* in_sizes, int n_in,
                              void* d_out, int out_size, void* d_ws, size_t ws_size,
                              hipStream_t stream) {
    const float* x    = (const float*)d_in[0];
    const int*   mask = (const int*)  d_in[1];
    const float* Wq   = (const float*)d_in[2];
    const float* bq   = (const float*)d_in[3];
    const float* Wk   = (const float*)d_in[4];
    const float* bk   = (const float*)d_in[5];
    const float* Wv   = (const float*)d_in[6];
    const float* bv   = (const float*)d_in[7];
    const float* Wo   = (const float*)d_in[8];
    const float* bo   = (const float*)d_in[9];
    float* out = (float*)d_out;

    char*  ws  = (char*)d_ws;
    size_t off = 0;
    auto alloc = [&](size_t bytes) {
        char* p = ws + off;
        off = (off + bytes + 255) & ~(size_t)255;
        return p;
    };
    const size_t headBytes = (size_t)B_ * H_ * S_ * DK_ * 2;  // 12.58 MB
    unsigned short* xb  = (unsigned short*)alloc((size_t)MTOT * D_ * 2);
    unsigned short* wqb = (unsigned short*)alloc((size_t)D_ * D_ * 2);
    unsigned short* wkb = (unsigned short*)alloc((size_t)D_ * D_ * 2);
    unsigned short* wvb = (unsigned short*)alloc((size_t)D_ * D_ * 2);
    unsigned short* wob = (unsigned short*)alloc((size_t)D_ * D_ * 2);
    unsigned short* Qh  = (unsigned short*)alloc(headBytes);
    unsigned short* Kh  = (unsigned short*)alloc(headBytes);
    unsigned short* Vt  = (unsigned short*)alloc(headBytes);
    unsigned short* Oc  = (unsigned short*)alloc((size_t)MTOT * D_ * 2);

    // fp32 -> bf16
    cvt_f32_bf16<<<(MTOT * D_) / 256, 256, 0, stream>>>(x,  xb,  MTOT * D_);
    cvt_f32_bf16<<<(D_ * D_)   / 256, 256, 0, stream>>>(Wq, wqb, D_ * D_);
    cvt_f32_bf16<<<(D_ * D_)   / 256, 256, 0, stream>>>(Wk, wkb, D_ * D_);
    cvt_f32_bf16<<<(D_ * D_)   / 256, 256, 0, stream>>>(Wv, wvb, D_ * D_);
    cvt_f32_bf16<<<(D_ * D_)   / 256, 256, 0, stream>>>(Wo, wob, D_ * D_);

    // Q/K/V projections (bias fused, head-split / V-transposed stores fused)
    dim3 gg(MTOT / 128, D_ / 128);
    gemm_wmma<<<gg, 256, 0, stream>>>(xb, wqb, bq, Qh, nullptr, 0);
    gemm_wmma<<<gg, 256, 0, stream>>>(xb, wkb, bk, Kh, nullptr, 1);
    gemm_wmma<<<gg, 256, 0, stream>>>(xb, wvb, bv, Vt, nullptr, 2);

    // flash attention: (S/128) x H x B blocks, 4 waves each
    dim3 ga(S_ / 128, H_, B_);
    attn_wmma<<<ga, 128, 0, stream>>>(Qh, Kh, Vt, mask, Oc);

    // output projection -> fp32 d_out
    gemm_wmma<<<gg, 256, 0, stream>>>(Oc, wob, bo, nullptr, out, 3);
}
